// SigLipEncoderLayerTome_64398739636964
// MI455X (gfx1250) — matrix operand order repaired and checked by reading, not compiled, not run on hardware
//
#include <hip/hip_runtime.h>
#include <hip/hip_bf16.h>
#include <stdint.h>

// ---------------- problem constants ----------------
#define B_    8
#define N_    729
#define C_    1152
#define H_    16
#define D_    72
#define F_    4304
#define R_    23
#define NM_   706            // N - R
#define NA_   365            // even tokens  (a side)
#define NB_   364            // odd tokens   (b side)
#define UNM_  342            // NA_ - R_
#define NKT_  46             // ceil(N/16) key tiles
#define NP_   736            // padded key/query count (46*16)
#define DP_   96             // per-head D padded to 3*32
#define FP_   4352           // F padded to 34*128 (also mult of 32)
#define MBP_  5840           // B*N padded to 365*16
#define MB_   5832           // B*N
#define MM_   5648           // B*NM (= 353*16 exactly)
#define SCALE_ 0.1178511301977579f   // 72^-0.5

typedef __bf16 bf16;
typedef __bf16 v16bf __attribute__((ext_vector_type(16)));
typedef __bf16 v8bf  __attribute__((ext_vector_type(8)));
typedef float  v8f   __attribute__((ext_vector_type(8)));

// WMMA fragment layouts used here (wave32, 16-bit; k order is this file's choice, same for A/B):
//  A (16x32 chunk): lane row m = lane&15 ; elements t -> k = kgrp*16 + t
//                   (contig 16, matching B)  (kgrp = lane>>4)
//  B (32x16 chunk): lane col n = lane&15 ; elements t -> k = kgrp*16 + t (contig 16)
//  C/D (16x16):     acc[i] -> m = i + kgrp*8, n = lane&15
__device__ __forceinline__ v16bf mk16(v8bf lo, v8bf hi) {
    v16bf r;
#pragma unroll
    for (int t = 0; t < 8; t++) { r[t] = lo[t]; r[t + 8] = hi[t]; }
    return r;
}

__device__ __forceinline__ float gelu_tanh(float x) {
    float x3 = x * x * x;
    float t  = tanhf(0.7978845608028654f * (x + 0.044715f * x3));
    return 0.5f * x * (1.f + t);
}

// ---------------- zero fill (16B granules) ----------------
__global__ void k_fill0(uint4* __restrict__ p, long n) {
    for (long i = (long)blockIdx.x * 256 + threadIdx.x; i < n; i += (long)gridDim.x * 256)
        p[i] = make_uint4(0u, 0u, 0u, 0u);
}

// ---------------- f32 weight -> bf16 transposed+padded ----------------
// out[n][k] (NcPad x KPad) = w[k][n] (Krows x Ncols), zero pad
__global__ void k_cvt_t(const float* __restrict__ w, bf16* __restrict__ out,
                        int Krows, int Ncols, int KPad, long total) {
    for (long i = (long)blockIdx.x * 256 + threadIdx.x; i < total; i += (long)gridDim.x * 256) {
        int k = (int)(i % KPad);
        int n = (int)(i / KPad);
        bf16 v = (bf16)0.f;
        if (k < Krows && n < Ncols) v = (bf16)w[(size_t)k * Ncols + n];
        out[i] = v;
    }
}

// ---------------- bias pad ----------------
__global__ void k_pad_f32(const float* __restrict__ in, float* __restrict__ out,
                          int n_in, int n_out) {
    int i = blockIdx.x * 256 + threadIdx.x;
    if (i < n_out) out[i] = (i < n_in) ? in[i] : 0.f;
}

// ---------------- mask pad: [B][NP][NP], -1e30 in pads ----------------
__global__ void k_pad_mask(const float* __restrict__ mask, float* __restrict__ mp, long total) {
    for (long i = (long)blockIdx.x * 256 + threadIdx.x; i < total; i += (long)gridDim.x * 256) {
        int key = (int)(i % NP_);
        long t  = i / NP_;
        int qr  = (int)(t % NP_);
        int b   = (int)(t / NP_);
        float v = -1e30f;
        if (qr < N_ && key < N_) v = mask[((size_t)b * N_ + qr) * N_ + key];
        mp[i] = v;
    }
}

// ---------------- LayerNorm (one row per block, 256 thr) ----------------
__global__ __launch_bounds__(256) void k_ln(const float* __restrict__ x,
                                            const float* __restrict__ g,
                                            const float* __restrict__ be,
                                            bf16* __restrict__ out, int C) {
    int row = blockIdx.x;
    const float* xr = x + (size_t)row * C;
    __shared__ float red0[8], red1[8];
    float s = 0.f, s2 = 0.f;
    for (int c = threadIdx.x; c < C; c += 256) { float v = xr[c]; s += v; s2 += v * v; }
    for (int o = 16; o > 0; o >>= 1) { s += __shfl_xor(s, o); s2 += __shfl_xor(s2, o); }
    int w = threadIdx.x >> 5, l = threadIdx.x & 31;
    if (l == 0) { red0[w] = s; red1[w] = s2; }
    __syncthreads();
    if (w == 0) {
        s  = (l < 8) ? red0[l] : 0.f;
        s2 = (l < 8) ? red1[l] : 0.f;
        for (int o = 4; o > 0; o >>= 1) { s += __shfl_xor(s, o); s2 += __shfl_xor(s2, o); }
        if (l == 0) { red0[0] = s; red1[0] = s2; }
    }
    __syncthreads();
    float mean = red0[0] / (float)C;
    float var  = red1[0] / (float)C - mean * mean;
    float rstd = rsqrtf(var + 1e-6f);
    for (int c = threadIdx.x; c < C; c += 256)
        out[(size_t)row * C + c] = (bf16)((xr[c] - mean) * rstd * g[c] + be[c]);
}

// ---------------- generic bf16 WMMA GEMM, transposed-B, fused epilogue ----
// out = A[MPad,K] @ Bt[NcPad,K]^T + bias ; K, NcPad multiples of 32/128.
// SMODE: 0 plain [*,Nc] (OUT_BF16 picks type), 1 headpad bf16 [B][NP][H][DP],
//        2 head-transposed bf16 [B][H][DP][NP]
template <int ACT, int SMODE, int OUT_BF16, int ADD_RES>
__global__ __launch_bounds__(256) void k_gemm(const bf16* __restrict__ A,
                                              const bf16* __restrict__ Bt,
                                              const float* __restrict__ bias,
                                              const float* __restrict__ res,
                                              void* __restrict__ outp,
                                              int Mreal, int K, int Nc) {
    __shared__ bf16 aT[16][40];
    const int wv = threadIdx.x >> 5, lane = threadIdx.x & 31;
    const int mloc = lane & 15, kgrp = lane >> 4;
    const int row0 = blockIdx.y * 16;
    const int ncol = blockIdx.x * 128 + wv * 16 + mloc;
    const size_t brow = (size_t)ncol * K;
    v8f acc = {};

    for (int kc = 0; kc < K; kc += 32) {
        if (threadIdx.x < 64) {                    // A tile 16x32 via 16B stores
            int r = threadIdx.x >> 2, q4 = threadIdx.x & 3;
            *(v8bf*)&aT[r][q4 * 8] =
                *(const v8bf*)&A[(size_t)(row0 + r) * K + kc + q4 * 8];
        }
        __syncthreads();
        v8bf alo = *(const v8bf*)&aT[mloc][kgrp * 16];
        v8bf ahi = *(const v8bf*)&aT[mloc][kgrp * 16 + 8];
        v8bf blo = *(const v8bf*)&Bt[brow + kc + kgrp * 16];
        v8bf bhi = *(const v8bf*)&Bt[brow + kc + kgrp * 16 + 8];
        if (kc + 32 < K) __builtin_prefetch(&Bt[brow + kc + 32 + kgrp * 16], 0, 3);
        acc = __builtin_amdgcn_wmma_f32_16x16x32_bf16(false, mk16(alo, ahi),
                                                      false, mk16(blo, bhi),
                                                      (short)0, acc, false, false);
        __syncthreads();
    }

    const int hh = ncol / D_, dd = ncol - hh * D_;   // used by SMODE 1/2
#pragma unroll
    for (int i = 0; i < 8; i++) {
        int gm = row0 + i + (kgrp << 3);
        if (gm >= Mreal) continue;
        float vo = acc[i] + bias[ncol];
        if (ACT == 1) vo = gelu_tanh(vo);
        if (ADD_RES)  vo += res[(size_t)gm * Nc + ncol];
        if (SMODE == 0) {
            if (OUT_BF16) ((bf16*)outp)[(size_t)gm * Nc + ncol] = (bf16)vo;
            else          ((float*)outp)[(size_t)gm * Nc + ncol] = vo;
        } else {
            int bb = gm / N_, nn = gm - bb * N_;
            if (SMODE == 1)
                ((bf16*)outp)[(((size_t)bb * NP_ + nn) * H_ + hh) * DP_ + dd] = (bf16)vo;
            else
                ((bf16*)outp)[(((size_t)bb * H_ + hh) * DP_ + dd) * NP_ + nn] = (bf16)vo;
        }
    }
}

// ---------------- fused attention: QK^T -> softmax -> AV ----------------
// qp/kp: [B][NP][H][DP] bf16 (zero padded) ; vT: [B][H][DP][NP] bf16
// maskp: [B][NP][NP] f32 (-1e30 pads) ; out ob: [MBP][C] bf16 plain
__global__ __launch_bounds__(32) void k_attn(const bf16* __restrict__ qp,
                                             const bf16* __restrict__ kp,
                                             const bf16* __restrict__ vT,
                                             const float* __restrict__ maskp,
                                             bf16* __restrict__ ob) {
    const int mt = blockIdx.x, h = blockIdx.y, b = blockIdx.z;
    const int lane = threadIdx.x;
    const int mloc = lane & 15, kgrp = lane >> 4;
    __shared__ bf16 sc[16][NP_ + 8];

    // Q fragments, D padded to 96 (3 chunks of 32)
    const int qrow = mt * 16 + mloc;
    const size_t qbase = (((size_t)b * NP_ + qrow) * H_ + h) * DP_;
    v16bf aq[3];
#pragma unroll
    for (int c = 0; c < 3; c++)
        aq[c] = mk16(*(const v8bf*)&qp[qbase + c * 32 + kgrp * 16],
                     *(const v8bf*)&qp[qbase + c * 32 + kgrp * 16 + 8]);

    // scores -> bf16 LDS
    for (int nt = 0; nt < NKT_; nt++) {
        const int keyn = nt * 16 + mloc;
        const size_t kbase = (((size_t)b * NP_ + keyn) * H_ + h) * DP_;
        v8f acc = {};
#pragma unroll
        for (int c = 0; c < 3; c++) {
            v16bf bk = mk16(*(const v8bf*)&kp[kbase + c * 32 + kgrp * 16],
                            *(const v8bf*)&kp[kbase + c * 32 + kgrp * 16 + 8]);
            acc = __builtin_amdgcn_wmma_f32_16x16x32_bf16(false, aq[c], false, bk,
                                                          (short)0, acc, false, false);
        }
#pragma unroll
        for (int i = 0; i < 8; i++) {
            int m  = i + (kgrp << 3);
            int qr = mt * 16 + m;
            int ky = nt * 16 + mloc;
            float s = acc[i] * SCALE_ + maskp[((size_t)b * NP_ + qr) * NP_ + ky];
            sc[m][ky] = (bf16)s;
        }
    }
    __syncthreads();

    // row softmax over NP_ (bf16 in LDS; pads are -1e30 -> exp 0)
    if (lane < 16) {
        float mx = -1e30f;
        for (int c2 = 0; c2 < NP_; c2++) mx = fmaxf(mx, (float)sc[lane][c2]);
        float sum = 0.f;
        for (int c2 = 0; c2 < NP_; c2++) {
            float e = __expf((float)sc[lane][c2] - mx);
            sc[lane][c2] = (bf16)e; sum += e;
        }
        float inv = 1.f / sum;
        for (int c2 = 0; c2 < NP_; c2++)
            sc[lane][c2] = (bf16)((float)sc[lane][c2] * inv);
    }
    __syncthreads();

    // out = P @ V  (5 col tiles of 16 over DP, 23 K-chunks of 32 over NP)
    for (int ot = 0; ot < 5; ot++) {
        const int dcol = ot * 16 + mloc;
        const size_t vbase = (((size_t)b * H_ + h) * DP_ + dcol) * NP_;
        v8f acc = {};
        for (int kt = 0; kt < 23; kt++) {
            v16bf ap = mk16(*(const v8bf*)&sc[mloc][kt * 32 + kgrp * 16],
                            *(const v8bf*)&sc[mloc][kt * 32 + kgrp * 16 + 8]);
            v16bf bv = mk16(*(const v8bf*)&vT[vbase + kt * 32 + kgrp * 16],
                            *(const v8bf*)&vT[vbase + kt * 32 + kgrp * 16 + 8]);
            acc = __builtin_amdgcn_wmma_f32_16x16x32_bf16(false, ap, false, bv,
                                                          (short)0, acc, false, false);
        }
#pragma unroll
        for (int i = 0; i < 8; i++) {
            int qr = mt * 16 + i + (kgrp << 3);
            if (qr < N_ && dcol < D_)
                ob[((size_t)b * N_ + qr) * C_ + h * D_ + dcol] = (bf16)acc[i];
        }
    }
}

// ---------------- ToMe: metric = mean_h(k), L2-normalized ----------------
__global__ __launch_bounds__(128) void k_metric(const bf16* __restrict__ kp,
                                                float* __restrict__ mnorm) {
    int row = blockIdx.x;                 // b*N + n
    int bb = row / N_, nn = row - bb * N_;
    __shared__ float buf[D_];
    __shared__ float rn;
    int t = threadIdx.x;
    float s = 0.f;
    if (t < D_) {
        for (int h = 0; h < H_; h++)
            s += (float)kp[(((size_t)bb * NP_ + nn) * H_ + h) * DP_ + t];
        s *= (1.f / (float)H_);
        buf[t] = s;
    }
    __syncthreads();
    if (t == 0) {
        float qv = 0.f;
        for (int i = 0; i < D_; i++) qv += buf[i] * buf[i];
        rn = rsqrtf(qv + 1e-30f);
    }
    __syncthreads();
    if (t < D_) mnorm[(size_t)row * D_ + t] = s * rn;
}

// ---------------- ToMe: per-even-token best odd match ----------------
__global__ __launch_bounds__(32) void k_tome_score(const float* __restrict__ mnorm,
                                                   float* __restrict__ node_max,
                                                   int* __restrict__ node_idx) {
    int i = blockIdx.x, b = blockIdx.y, lane = threadIdx.x;
    const float* a = mnorm + ((size_t)b * N_ + 2 * i) * D_;
    float best = -1e30f; int bidx = 0x7fffffff;
    for (int j = lane; j < NB_; j += 32) {
        const float* bm = mnorm + ((size_t)b * N_ + 2 * j + 1) * D_;
        float d = 0.f;
#pragma unroll 8
        for (int qd = 0; qd < D_; qd++) d += a[qd] * bm[qd];
        if (d > best || (d == best && j < bidx)) { best = d; bidx = j; }
    }
    for (int o = 16; o > 0; o >>= 1) {
        float ov = __shfl_xor(best, o);
        int   oi = __shfl_xor(bidx, o);
        if (ov > best || (ov == best && oi < bidx)) { best = ov; bidx = oi; }
    }
    if (lane == 0) { node_max[b * NA_ + i] = best; node_idx[b * NA_ + i] = bidx; }
}

// ---------------- ToMe: stable descending argsort + plan ----------------
__global__ __launch_bounds__(256) void k_tome_plan(const float* __restrict__ node_max,
                                                   const int* __restrict__ node_idx,
                                                   int* __restrict__ edge_idx,
                                                   int* __restrict__ src_idx,
                                                   int* __restrict__ dst_idx,
                                                   int* __restrict__ counts) {
    int b = blockIdx.x, t = threadIdx.x;
    const float* nm = node_max + b * NA_;
    for (int i = t; i < NA_; i += 256) {
        float vi = nm[i]; int r = 0;
        for (int j = 0; j < NA_; j++) {
            float vj = nm[j];
            if (vj > vi || (vj == vi && j < i)) r++;
        }
        edge_idx[b * NA_ + r] = i;
    }
    for (int j = t; j < NB_; j += 256) counts[b * NB_ + j] = 1;
    __syncthreads();
    if (t < R_) {
        int s = edge_idx[b * NA_ + t];
        src_idx[b * R_ + t] = s;
        int d = node_idx[b * NA_ + s];
        dst_idx[b * R_ + t] = d;
        atomicAdd(&counts[b * NB_ + d], 1);
    }
}

// ---------------- ToMe: weighted-average merge ----------------
__global__ __launch_bounds__(256) void k_tome_merge(const float* __restrict__ hfull,
                                                    const int* __restrict__ edge_idx,
                                                    const int* __restrict__ src_idx,
                                                    const int* __restrict__ dst_idx,
                                                    const int* __restrict__ counts,
                                                    float* __restrict__ hm) {
    int o = blockIdx.x, b = blockIdx.y;
    for (int c = threadIdx.x; c < C_; c += 256) {
        float acc;
        if (o < UNM_) {
            int tok = 2 * edge_idx[b * NA_ + R_ + o];
            acc = hfull[((size_t)b * N_ + tok) * C_ + c];
        } else {
            int j = o - UNM_;
            acc = hfull[((size_t)b * N_ + 2 * j + 1) * C_ + c];
            for (int p = 0; p < R_; p++)
                if (dst_idx[b * R_ + p] == j)
                    acc += hfull[((size_t)b * N_ + 2 * src_idx[b * R_ + p]) * C_ + c];
            acc /= (float)counts[b * NB_ + j];
        }
        hm[((size_t)b * NM_ + o) * C_ + c] = acc;
    }
}

// ---------------- host-side launch ----------------
extern "C" void kernel_launch(void* const* d_in, const int* in_sizes, int n_in,
                              void* d_out, int out_size, void* d_ws, size_t ws_size,
                              hipStream_t stream) {
    (void)in_sizes; (void)n_in; (void)out_size; (void)ws_size;

    const float* hs    = (const float*)d_in[0];
    const float* mask  = (const float*)d_in[1];
    const float* wq    = (const float*)d_in[2];
    const float* bq    = (const float*)d_in[3];
    const float* wk    = (const float*)d_in[4];
    const float* bk    = (const float*)d_in[5];
    const float* wvp   = (const float*)d_in[6];
    const float* bv    = (const float*)d_in[7];
    const float* wo    = (const float*)d_in[8];
    const float* bo    = (const float*)d_in[9];
    const float* ln1w  = (const float*)d_in[10];
    const float* ln1b  = (const float*)d_in[11];
    const float* ln2w  = (const float*)d_in[12];
    const float* ln2b  = (const float*)d_in[13];
    const float* fc1w  = (const float*)d_in[14];
    const float* fc1b  = (const float*)d_in[15];
    const float* fc2w  = (const float*)d_in[16];
    const float* fc2b  = (const float*)d_in[17];

    uint8_t* wsp = (uint8_t*)d_ws;
    auto alloc = [&](size_t bytes) -> void* {
        void* p = (void*)wsp;
        wsp += (bytes + 255) & ~(size_t)255;
        return p;
    };
    const size_t HEADSZ = (size_t)B_ * NP_ * H_ * DP_;    // padded q/k/v elem count
    bf16* wtq  = (bf16*)alloc((size_t)C_ * C_ * 2);       // [Nc=1152][K=1152] transposed
    bf16* wtk  = (bf16*)alloc((size_t)C_ * C_ * 2);
    bf16* wtv  = (bf16*)alloc((size_t)C_ * C_ * 2);
    bf16* wto  = (bf16*)alloc((size_t)C_ * C_ * 2);
    bf16* wt1  = (bf16*)alloc((size_t)FP_ * C_ * 2);      // fc1^T [4352][1152]
    bf16* wt2  = (bf16*)alloc((size_t)C_ * FP_ * 2);      // fc2^T [1152][4352]
    float* b1p = (float*)alloc((size_t)FP_ * 4);
    bf16* xlnp = (bf16*)alloc((size_t)MBP_ * C_ * 2);
    bf16* qp   = (bf16*)alloc(HEADSZ * 2);
    bf16* kp   = (bf16*)alloc(HEADSZ * 2);
    bf16* vT   = (bf16*)alloc(HEADSZ * 2);
    bf16* ob   = (bf16*)alloc((size_t)MBP_ * C_ * 2);
    float* mkp = (float*)alloc((size_t)B_ * NP_ * NP_ * 4);
    float* hfull = (float*)alloc((size_t)MB_ * C_ * 4);
    float* mnorm = (float*)alloc((size_t)MB_ * D_ * 4);
    float* nmax  = (float*)alloc((size_t)B_ * NA_ * 4);
    int*   nidx  = (int*)  alloc((size_t)B_ * NA_ * 4);
    int*   eidx  = (int*)  alloc((size_t)B_ * NA_ * 4);
    int*   sidx  = (int*)  alloc((size_t)B_ * R_ * 4);
    int*   didx  = (int*)  alloc((size_t)B_ * R_ * 4);
    int*   cnts  = (int*)  alloc((size_t)B_ * NB_ * 4);
    float* hm    = (float*)alloc((size_t)MM_ * C_ * 4);
    bf16*  x2    = (bf16*) alloc((size_t)MM_ * C_ * 2);
    bf16*  h1    = (bf16*) alloc((size_t)MM_ * FP_ * 2);

    // zero-fill padded activation buffers (pads must be 0 for WMMA correctness)
    k_fill0<<<2048, 256, 0, stream>>>((uint4*)qp, (long)(HEADSZ * 2 / 16));
    k_fill0<<<2048, 256, 0, stream>>>((uint4*)kp, (long)(HEADSZ * 2 / 16));
    k_fill0<<<2048, 256, 0, stream>>>((uint4*)vT, (long)(HEADSZ * 2 / 16));
    k_fill0<<<8, 256, 0, stream>>>((uint4*)(xlnp + (size_t)MB_ * C_),
                                   (long)((size_t)(MBP_ - MB_) * C_ * 2 / 16));
    k_fill0<<<8, 256, 0, stream>>>((uint4*)(ob + (size_t)MB_ * C_),
                                   (long)((size_t)(MBP_ - MB_) * C_ * 2 / 16));

    // weights: transpose + pad -> bf16
    k_cvt_t<<<4096, 256, 0, stream>>>(wq,  wtq, C_, C_, C_, (long)C_ * C_);
    k_cvt_t<<<4096, 256, 0, stream>>>(wk,  wtk, C_, C_, C_, (long)C_ * C_);
    k_cvt_t<<<4096, 256, 0, stream>>>(wvp, wtv, C_, C_, C_, (long)C_ * C_);
    k_cvt_t<<<4096, 256, 0, stream>>>(wo,  wto, C_, C_, C_, (long)C_ * C_);
    k_cvt_t<<<8192, 256, 0, stream>>>(fc1w, wt1, C_, F_, C_, (long)FP_ * C_);
    k_cvt_t<<<8192, 256, 0, stream>>>(fc2w, wt2, F_, C_, FP_, (long)C_ * FP_);
    k_pad_f32<<<(FP_ + 255) / 256, 256, 0, stream>>>(fc1b, b1p, F_, FP_);
    k_pad_mask<<<4096, 256, 0, stream>>>(mask, mkp, (long)B_ * NP_ * NP_);

    // LN1
    k_ln<<<MB_, 256, 0, stream>>>(hs, ln1w, ln1b, xlnp, C_);

    // QKV projections into padded head layouts
    dim3 gP(9, MBP_ / 16);
    k_gemm<0, 1, 1, 0><<<gP, 256, 0, stream>>>(xlnp, wtq, bq, nullptr, qp, MB_, C_, C_);
    k_gemm<0, 1, 1, 0><<<gP, 256, 0, stream>>>(xlnp, wtk, bk, nullptr, kp, MB_, C_, C_);
    k_gemm<0, 2, 1, 0><<<gP, 256, 0, stream>>>(xlnp, wtv, bv, nullptr, vT, MB_, C_, C_);

    // metric (head-mean of k, L2-normalized)
    k_metric<<<MB_, 128, 0, stream>>>(kp, mnorm);

    // fused attention
    k_attn<<<dim3(NKT_, H_, B_), 32, 0, stream>>>(qp, kp, vT, mkp, ob);

    // output projection + residual -> hfull (f32)
    k_gemm<0, 0, 0, 1><<<gP, 256, 0, stream>>>(ob, wto, bo, hs, hfull, MB_, C_, C_);

    // ToMe
    k_tome_score<<<dim3(NA_, B_), 32, 0, stream>>>(mnorm, nmax, nidx);
    k_tome_plan<<<B_, 256, 0, stream>>>(nmax, nidx, eidx, sidx, didx, cnts);
    k_tome_merge<<<dim3(NM_, B_), 256, 0, stream>>>(hfull, eidx, sidx, didx, cnts, hm);

    // LN2
    k_ln<<<MM_, 256, 0, stream>>>(hm, ln2w, ln2b, x2, C_);

    // MLP
    k_gemm<1, 0, 1, 0><<<dim3(FP_ / 128, MM_ / 16), 256, 0, stream>>>(
        x2, wt1, b1p, nullptr, h1, MM_, C_, FP_);
    k_gemm<0, 0, 0, 1><<<dim3(9, MM_ / 16), 256, 0, stream>>>(
        h1, wt2, fc2b, hm, (float*)d_out, MM_, FP_, C_);
}
